// LSTMPCell_61710090109320
// MI455X (gfx1250) — matrix operand order repaired and checked by reading, not compiled, not run on hardware
//
#include <hip/hip_runtime.h>
#include <hip/hip_bf16.h>
#include <math.h>

// Problem dims (fixed by the reference)
#define B_ 8192
#define I_ 1024
#define H_ 2048
#define P_ 512

typedef __attribute__((ext_vector_type(16))) __bf16 v16bf;
typedef __attribute__((ext_vector_type(8)))  __bf16 v8bf;
typedef __attribute__((ext_vector_type(8)))  float  v8f;

// f32 -> bf16 bits, round-to-nearest-even
__device__ __forceinline__ unsigned bfbits(float x) {
  unsigned u = __builtin_bit_cast(unsigned, x);
  u += 0x7FFFu + ((u >> 16) & 1u);
  return u >> 16;
}
__device__ __forceinline__ __bf16 f2bf(float x) {
  unsigned short h = (unsigned short)bfbits(x);
  return __builtin_bit_cast(__bf16, h);
}

// ---------------------------------------------------------------------------
// One-pass f32 -> bf16 conversion (memory-bound; runs once per operand).
// ---------------------------------------------------------------------------
__global__ __launch_bounds__(256) void cvt_f32_bf16_kernel(
    const float* __restrict__ src, unsigned* __restrict__ dst, int n4) {
  int i = blockIdx.x * blockDim.x + threadIdx.x;
  if (i >= n4) return;
  float4 v = ((const float4*)src)[i];
  uint2 r;
  r.x = bfbits(v.x) | (bfbits(v.y) << 16);
  r.y = bfbits(v.z) | (bfbits(v.w) << 16);
  ((uint2*)dst)[i] = r;
}

// ---------------------------------------------------------------------------
// Fragment loaders from bf16 row-major sources (pure b128 loads, no VALU).
// A (16x32, ISA 7.12.2): lanes 0-15 e[0..7]=K0..7, e[8..15]=K16..23;
//                        lanes 16-31 e[0..7]=K8..15, e[8..15]=K24..31.
// B (32x16): same K layout as A (lanes 0-15 e[0..7]=K0..7, e[8..15]=K16..23; lanes 16-31 K8..15, K24..31).
// ---------------------------------------------------------------------------
__device__ __forceinline__ v16bf load_frag_a_bf(const __bf16* p, int half) {
  v8bf lo = *(const v8bf*)(p + half * 8);
  v8bf hi = *(const v8bf*)(p + 16 + half * 8);
  return __builtin_shufflevector(lo, hi, 0, 1, 2, 3, 4, 5, 6, 7,
                                 8, 9, 10, 11, 12, 13, 14, 15);
}
__device__ __forceinline__ v16bf load_frag_b_bf(const __bf16* p, int half) {
  v8bf lo = *(const v8bf*)(p + half * 8);
  v8bf hi = *(const v8bf*)(p + 16 + half * 8);
  return __builtin_shufflevector(lo, hi, 0, 1, 2, 3, 4, 5, 6, 7,
                                 8, 9, 10, 11, 12, 13, 14, 15);
}

#define WMMA_BF16(A, Bm, C) \
  __builtin_amdgcn_wmma_f32_16x16x32_bf16(false, (A), false, (Bm), (short)0, (C), false, false)

// ---------------------------------------------------------------------------
// Kernel 1: fused  gates = x@w_x^T + b_x + hx@w_h^T + b_h,
//           ct = f*cx + i*c -> d_out,  t = o*tanh(ct) -> bf16 workspace.
// Each wave: M=32 x N=16 per gate (2 A-frags, 8 accumulators) so every
// B-fragment feeds two WMMAs (12 b128 loads per 8 WMMAs).
// Grid: (H/128, B/32). Block: 256 threads = 8 waves side by side in N.
// ---------------------------------------------------------------------------
__global__ __launch_bounds__(256) void lstmp_gates_kernel(
    const __bf16* __restrict__ input, const __bf16* __restrict__ hx,
    const float* __restrict__ cx, const __bf16* __restrict__ wx,
    const float* __restrict__ bx, const __bf16* __restrict__ wh,
    const float* __restrict__ bh, float* __restrict__ ct_out,
    __bf16* __restrict__ t_ws) {
  const int tid  = threadIdx.x;
  const int lane = tid & 31;
  const int wave = tid >> 5;         // 0..7, tiles N
  const int half = lane >> 4;
  const int l    = lane & 15;
  const int m0 = blockIdx.y * 32;                 // batch rows (wave does 32)
  const int n0 = blockIdx.x * 128 + wave * 16;    // gate-local column in [0,H)

  v8f aI0 = {}, aI1 = {}, aF0 = {}, aF1 = {};
  v8f aC0 = {}, aC1 = {}, aO0 = {}, aO1 = {};

  // ---- input @ w_x^T : K = I_ ----
  {
    const __bf16* aRow0 = input + (size_t)(m0 + l) * I_;
    const __bf16* aRow1 = input + (size_t)(m0 + 16 + l) * I_;
    const __bf16* r0 = wx + (size_t)(0 * H_ + n0 + l) * I_;
    const __bf16* r1 = wx + (size_t)(1 * H_ + n0 + l) * I_;
    const __bf16* r2 = wx + (size_t)(2 * H_ + n0 + l) * I_;
    const __bf16* r3 = wx + (size_t)(3 * H_ + n0 + l) * I_;
    for (int k0 = 0; k0 < I_; k0 += 32) {
      v16bf a0 = load_frag_a_bf(aRow0 + k0, half);
      v16bf a1 = load_frag_a_bf(aRow1 + k0, half);
      v16bf w;
      w = load_frag_b_bf(r0 + k0, half);
      aI0 = WMMA_BF16(a0, w, aI0);  aI1 = WMMA_BF16(a1, w, aI1);
      w = load_frag_b_bf(r1 + k0, half);
      aF0 = WMMA_BF16(a0, w, aF0);  aF1 = WMMA_BF16(a1, w, aF1);
      w = load_frag_b_bf(r2 + k0, half);
      aC0 = WMMA_BF16(a0, w, aC0);  aC1 = WMMA_BF16(a1, w, aC1);
      w = load_frag_b_bf(r3 + k0, half);
      aO0 = WMMA_BF16(a0, w, aO0);  aO1 = WMMA_BF16(a1, w, aO1);
    }
  }
  // ---- hx @ w_h^T : K = P_ ----
  {
    const __bf16* aRow0 = hx + (size_t)(m0 + l) * P_;
    const __bf16* aRow1 = hx + (size_t)(m0 + 16 + l) * P_;
    const __bf16* r0 = wh + (size_t)(0 * H_ + n0 + l) * P_;
    const __bf16* r1 = wh + (size_t)(1 * H_ + n0 + l) * P_;
    const __bf16* r2 = wh + (size_t)(2 * H_ + n0 + l) * P_;
    const __bf16* r3 = wh + (size_t)(3 * H_ + n0 + l) * P_;
    for (int k0 = 0; k0 < P_; k0 += 32) {
      v16bf a0 = load_frag_a_bf(aRow0 + k0, half);
      v16bf a1 = load_frag_a_bf(aRow1 + k0, half);
      v16bf w;
      w = load_frag_b_bf(r0 + k0, half);
      aI0 = WMMA_BF16(a0, w, aI0);  aI1 = WMMA_BF16(a1, w, aI1);
      w = load_frag_b_bf(r1 + k0, half);
      aF0 = WMMA_BF16(a0, w, aF0);  aF1 = WMMA_BF16(a1, w, aF1);
      w = load_frag_b_bf(r2 + k0, half);
      aC0 = WMMA_BF16(a0, w, aC0);  aC1 = WMMA_BF16(a1, w, aC1);
      w = load_frag_b_bf(r3 + k0, half);
      aO0 = WMMA_BF16(a0, w, aO0);  aO1 = WMMA_BF16(a1, w, aO1);
    }
  }

  // ---- epilogue: bias, cell update, tanh, store ct & t ----
  const int n = n0 + l;                         // column in [0,H)
  const float bI = bx[0 * H_ + n] + bh[0 * H_ + n];
  const float bF = bx[1 * H_ + n] + bh[1 * H_ + n];
  const float bC = bx[2 * H_ + n] + bh[2 * H_ + n];
  const float bO = bx[3 * H_ + n] + bh[3 * H_ + n];
#pragma unroll
  for (int mh = 0; mh < 2; ++mh) {
    const v8f& vI = mh ? aI1 : aI0;
    const v8f& vF = mh ? aF1 : aF0;
    const v8f& vC = mh ? aC1 : aC0;
    const v8f& vO = mh ? aO1 : aO0;
#pragma unroll
    for (int j = 0; j < 8; ++j) {
      const int m = m0 + mh * 16 + j + half * 8;  // row in [0,B)
      const size_t idx = (size_t)m * H_ + n;
      const float ig = vI[j] + bI;
      const float fg = vF[j] + bF;
      const float cg = vC[j] + bC;
      const float og = vO[j] + bO;
      const float ctv = fg * cx[idx] + ig * cg;   // NO sigmoid/tanh on gates (faithful)
      ct_out[idx] = ctv;
      t_ws[idx] = f2bf(og * tanhf(ctv));
    }
  }
}

// ---------------------------------------------------------------------------
// Kernel 2: ht = t @ w_proj^T  (both operands bf16), M=32 per wave.
// Grid: (P/128, B/32). Block: 256 threads = 8 waves side by side in N.
// ---------------------------------------------------------------------------
__global__ __launch_bounds__(256) void lstmp_proj_kernel(
    const __bf16* __restrict__ t_ws, const __bf16* __restrict__ wproj,
    float* __restrict__ ht_out) {
  const int tid  = threadIdx.x;
  const int lane = tid & 31;
  const int wave = tid >> 5;
  const int half = lane >> 4;
  const int l    = lane & 15;
  const int m0 = blockIdx.y * 32;
  const int n0 = blockIdx.x * 128 + wave * 16;    // projection column in [0,P)

  v8f acc0 = {}, acc1 = {};
  const __bf16* aRow0 = t_ws + (size_t)(m0 + l) * H_;
  const __bf16* aRow1 = t_ws + (size_t)(m0 + 16 + l) * H_;
  const __bf16* bRow  = wproj + (size_t)(n0 + l) * H_;
  for (int k0 = 0; k0 < H_; k0 += 32) {
    v16bf a0 = load_frag_a_bf(aRow0 + k0, half);
    v16bf a1 = load_frag_a_bf(aRow1 + k0, half);
    v16bf w  = load_frag_b_bf(bRow + k0, half);
    acc0 = WMMA_BF16(a0, w, acc0);
    acc1 = WMMA_BF16(a1, w, acc1);
  }
#pragma unroll
  for (int mh = 0; mh < 2; ++mh) {
    const v8f& v = mh ? acc1 : acc0;
#pragma unroll
    for (int j = 0; j < 8; ++j) {
      const int m = m0 + mh * 16 + j + half * 8;
      ht_out[(size_t)m * P_ + (n0 + l)] = v[j];
    }
  }
}

extern "C" void kernel_launch(void* const* d_in, const int* in_sizes, int n_in,
                              void* d_out, int out_size, void* d_ws, size_t ws_size,
                              hipStream_t stream) {
  (void)in_sizes; (void)n_in; (void)out_size; (void)ws_size;
  const float* input = (const float*)d_in[0];
  const float* hx    = (const float*)d_in[1];
  const float* cx    = (const float*)d_in[2];
  const float* wx    = (const float*)d_in[3];
  const float* bx    = (const float*)d_in[4];
  const float* wh    = (const float*)d_in[5];
  const float* bh    = (const float*)d_in[6];
  const float* wproj = (const float*)d_in[7];

  float* ht = (float*)d_out;                       // [B, P]  (first output)
  float* ct = (float*)d_out + (size_t)B_ * P_;     // [B, H]  (second output)

  // Workspace layout: input_bf 16MB | hx_bf 8MB | wx_bf 16MB | wh_bf 8MB |
  //                   wproj_bf 2MB | t 32MB   (total 82 MB)
  char* ws = (char*)d_ws;
  __bf16* input_bf = (__bf16*)(ws);
  __bf16* hx_bf    = (__bf16*)(ws + (size_t)B_ * I_ * 2);
  __bf16* wx_bf    = (__bf16*)(ws + (size_t)(B_ * I_ + B_ * P_) * 2);
  __bf16* wh_bf    = (__bf16*)(ws + (size_t)(B_ * I_ + B_ * P_ + 4 * H_ * I_) * 2);
  __bf16* wproj_bf = (__bf16*)(ws + (size_t)(B_ * I_ + B_ * P_ + 4 * H_ * I_ + 4 * H_ * P_) * 2);
  __bf16* t_ws     = (__bf16*)(ws + (size_t)(B_ * I_ + B_ * P_ + 4 * H_ * I_ + 4 * H_ * P_ + P_ * H_) * 2);

  dim3 blk(256, 1, 1);

  // One-pass conversions (memory-bound, ~7 us total at 23.3 TB/s).
  {
    int n4;
    n4 = (B_ * I_) / 4;
    cvt_f32_bf16_kernel<<<dim3((n4 + 255) / 256), blk, 0, stream>>>(input, (unsigned*)input_bf, n4);
    n4 = (B_ * P_) / 4;
    cvt_f32_bf16_kernel<<<dim3((n4 + 255) / 256), blk, 0, stream>>>(hx, (unsigned*)hx_bf, n4);
    n4 = (4 * H_ * I_) / 4;
    cvt_f32_bf16_kernel<<<dim3((n4 + 255) / 256), blk, 0, stream>>>(wx, (unsigned*)wx_bf, n4);
    n4 = (4 * H_ * P_) / 4;
    cvt_f32_bf16_kernel<<<dim3((n4 + 255) / 256), blk, 0, stream>>>(wh, (unsigned*)wh_bf, n4);
    n4 = (P_ * H_) / 4;
    cvt_f32_bf16_kernel<<<dim3((n4 + 255) / 256), blk, 0, stream>>>(wproj, (unsigned*)wproj_bf, n4);
  }

  dim3 g1(H_ / 128, B_ / 32, 1);                   // 16 x 256 workgroups
  lstmp_gates_kernel<<<g1, blk, 0, stream>>>(input_bf, hx_bf, cx, wx_bf, bx, wh_bf, bh, ct, t_ws);
  dim3 g2(P_ / 128, B_ / 32, 1);                   // 4 x 256 workgroups
  lstmp_proj_kernel<<<g2, blk, 0, stream>>>(t_ws, wproj_bf, ht);
}
